// MultiHeadAttentionDeBERTa_57552561766509
// MI455X (gfx1250) — matrix operand order repaired and checked by reading, not compiled, not run on hardware
//
#include <hip/hip_runtime.h>
#include <math.h>

#define B_   2
#define S_   2048
#define H_   1024
#define NH_  16
#define HD_  64
#define WIN_ 512
#define INV_SCALE 0.07216878364870323f  /* 1/sqrt(64*3) */

typedef unsigned short u16b;
typedef __attribute__((ext_vector_type(16))) __bf16    v16bf;
typedef __attribute__((ext_vector_type(8)))  float     v8f;
typedef __attribute__((ext_vector_type(4)))  float     v4f;
typedef __attribute__((ext_vector_type(4)))  unsigned  v4u;
typedef __attribute__((ext_vector_type(2)))  unsigned  v2u;

union FragU { v4u q[2]; v16bf v; };

__device__ inline u16b f2bf(float f) {
  union { float f; unsigned u; } x; x.f = f;
  unsigned r = (x.u + 0x7FFFu + ((x.u >> 16) & 1u)) >> 16;
  return (u16b)r;
}
__device__ inline float bf2f(u16b s) {
  union { unsigned u; float f; } x; x.u = ((unsigned)s) << 16;
  return x.f;
}

// A fragment 16x32 bf16 from per-lane pointer: 16 contiguous values
__device__ inline v16bf frag_load_a(const u16b* p) {
  FragU f;
  f.q[0] = *(const v4u*)p;
  f.q[1] = *(const v4u*)(p + 8);
  return f.v;
}
// B fragment 32x16 bf16 from per-lane pointer: 16 contiguous values
__device__ inline v16bf frag_load_b(const u16b* p) {
  FragU f;
  f.q[0] = *(const v4u*)p;
  f.q[1] = *(const v4u*)(p + 8);
  return f.v;
}
// convenience wrappers (lane-relative addressing computed inside)
__device__ inline v16bf frag_a(const u16b* base, int ld, int lane) {
  int h = lane >> 4, m = lane & 15;
  return frag_load_a(base + (long long)m * ld + 16 * h);
}
__device__ inline v16bf frag_b(const u16b* base, int ld, int lane) {
  int h = lane >> 4, n = lane & 15;
  return frag_load_b(base + (long long)n * ld + 16 * h);
}
__device__ inline v8f wmma_bf16(v16bf a, v16bf b, v8f c) {
  return __builtin_amdgcn_wmma_f32_16x16x32_bf16(false, a, false, b, (short)0, c,
                                                 false, false);
}

// ---------------- conversion / transpose / layernorm ----------------

// vectorized f32 -> bf16, 4 elements per thread
__global__ void f2bf_kernel(const float* __restrict__ in, u16b* __restrict__ out, int n4) {
  int i = blockIdx.x * 256 + threadIdx.x;
  if (i < n4) {
    v4f v = ((const v4f*)in)[i];
    v2u o;
    o.x = (unsigned)f2bf(v.x) | ((unsigned)f2bf(v.y) << 16);
    o.y = (unsigned)f2bf(v.z) | ((unsigned)f2bf(v.w) << 16);
    ((v2u*)out)[i] = o;
  }
}

// Wt[n*1024 + k] = bf16(W[k*1024 + n]); coalesced via 32x32 LDS tile
__global__ void wtrans_kernel(const float* __restrict__ W, u16b* __restrict__ Wt) {
  __shared__ u16b tile[32][34];
  int tx = threadIdx.x & 31, ty = threadIdx.x >> 5;    // 32 x 8
  int bx = blockIdx.x, by = blockIdx.y;                // n-block, k-block
#pragma unroll
  for (int r = 0; r < 4; ++r) {
    int k = by * 32 + ty + r * 8;
    tile[ty + r * 8][tx] = f2bf(W[(long long)k * H_ + bx * 32 + tx]);  // tile[k][n]
  }
  __syncthreads();
#pragma unroll
  for (int r = 0; r < 4; ++r) {
    int n = bx * 32 + ty + r * 8;
    Wt[(long long)n * H_ + by * 32 + tx] = tile[tx][ty + r * 8];
  }
}

// vT[((b*NH+h)*HD+d)*S + s] = v[(b*S+s)*H + h*HD + d]; 32x32 LDS tile per (b,h)
__global__ void vtrans_kernel(const u16b* __restrict__ v, u16b* __restrict__ vT) {
  __shared__ u16b tile[32][34];
  int tx = threadIdx.x & 31, ty = threadIdx.x >> 5;    // 32 x 8
  int s0 = blockIdx.x * 32, d0 = blockIdx.y * 32;
  int h = blockIdx.z & (NH_ - 1), b = blockIdx.z >> 4;
  const u16b* vz = v + (long long)b * S_ * H_ + h * HD_;
#pragma unroll
  for (int r = 0; r < 4; ++r)
    tile[ty + r * 8][tx] = vz[(long long)(s0 + ty + r * 8) * H_ + d0 + tx];  // tile[s][d]
  __syncthreads();
  u16b* vTz = vT + ((long long)(b * NH_ + h)) * HD_ * S_;
#pragma unroll
  for (int r = 0; r < 4; ++r)
    vTz[(long long)(d0 + ty + r * 8) * S_ + s0 + tx] = tile[tx][ty + r * 8];
}

__global__ void lnorm_kernel(const float* __restrict__ tbl,
                             const float* __restrict__ gamma,
                             const float* __restrict__ beta,
                             u16b* __restrict__ e) {
  __shared__ float s1[256], s2[256];
  int row = blockIdx.x, tid = threadIdx.x;
  const float* r = tbl + (long long)row * H_;
  float sum = 0.f, sq = 0.f;
  for (int c = tid; c < H_; c += 256) { float v = r[c]; sum += v; sq += v * v; }
  s1[tid] = sum; s2[tid] = sq; __syncthreads();
  for (int off = 128; off > 0; off >>= 1) {
    if (tid < off) { s1[tid] += s1[tid + off]; s2[tid] += s2[tid + off]; }
    __syncthreads();
  }
  float mu  = s1[0] * (1.f / H_);
  float var = s2[0] * (1.f / H_) - mu * mu;
  float rstd = rsqrtf(var + 1e-5f);
  for (int c = tid; c < H_; c += 256)
    e[(long long)row * H_ + c] = f2bf((r[c] - mu) * rstd * gamma[c] + beta[c]);
}

// ---------- generic batched WMMA GEMM (ping-pong double buffer, K%64==0) ----------
// C[m,n] = sum_k A[m,k] * Bt[n,k] (+bias[n]).  z = outer*innerCnt + inner.
__global__ __launch_bounds__(256)
void gemm_wmma(const u16b* __restrict__ A, int lda, long long aO, long long aI,
               const u16b* __restrict__ Bt, int ldb, long long bO, long long bI,
               const float* __restrict__ bias,
               void* __restrict__ C, int ldc, long long cO, long long cI,
               int innerCnt, int M, int N, int K, int cF32) {
  int lane = threadIdx.x & 31;
  int wave = threadIdx.x >> 5;
  int zo = blockIdx.z / innerCnt, zi = blockIdx.z % innerCnt;
  A  += zo * aO + zi * aI;
  Bt += zo * bO + zi * bI;
  int row0 = blockIdx.x * 128 + wave * 16;
  int col0 = blockIdx.y * 64;
  if (row0 >= M || col0 >= N) return;

  int h = lane >> 4, nl = lane & 15;
  // per-lane fragment base pointers (computed once; loop indexes with k0 only)
  const u16b* pA = A + (long long)(row0 + nl) * lda + 16 * h;
  const u16b* pB[4];
#pragma unroll
  for (int t = 0; t < 4; ++t)
    pB[t] = Bt + (long long)(col0 + t * 16 + nl) * ldb + 16 * h;

  v8f acc[4] = {};
  v16bf af[2];
  v16bf bb[2][4];
  af[0] = frag_load_a(pA);
#pragma unroll
  for (int t = 0; t < 4; ++t) bb[0][t] = frag_load_b(pB[t]);

  // K is a multiple of 64 for every call in this file (1024 or 64)
  for (int k0 = 0; k0 < K; k0 += 64) {
    int k1 = k0 + 32;
    af[1] = frag_load_a(pA + k1);
#pragma unroll
    for (int t = 0; t < 4; ++t) bb[1][t] = frag_load_b(pB[t] + k1);
#pragma unroll
    for (int t = 0; t < 4; ++t) acc[t] = wmma_bf16(af[0], bb[0][t], acc[t]);

    int k2 = (k0 + 64 < K) ? (k0 + 64) : 0;   // last iter: dummy (valid) address
    af[0] = frag_load_a(pA + k2);
#pragma unroll
    for (int t = 0; t < 4; ++t) bb[0][t] = frag_load_b(pB[t] + k2);
#pragma unroll
    for (int t = 0; t < 4; ++t) acc[t] = wmma_bf16(af[1], bb[1][t], acc[t]);
  }

  if (cF32) {
    float* Cp = (float*)C + zo * cO + zi * cI;
#pragma unroll
    for (int t = 0; t < 4; ++t) {
      int n = col0 + t * 16 + nl;
      float bv = bias ? bias[n] : 0.f;
#pragma unroll
      for (int i = 0; i < 8; ++i)
        Cp[(long long)(row0 + i + 8 * h) * ldc + n] = acc[t][i] + bv;
    }
  } else {
    u16b* Cp = (u16b*)C + zo * cO + zi * cI;
#pragma unroll
    for (int t = 0; t < 4; ++t) {
      int n = col0 + t * 16 + nl;
      float bv = bias ? bias[n] : 0.f;
#pragma unroll
      for (int i = 0; i < 8; ++i)
        Cp[(long long)(row0 + i + 8 * h) * ldc + n] = f2bf(acc[t][i] + bv);
    }
  }
}

// ---------------- flash attention with DeBERTa rel-pos ----------------
// one wave per (b, h, 16-query tile); streams 32-key tiles causally.
__global__ __launch_bounds__(32)
void attn_kernel(const u16b* __restrict__ qb, const u16b* __restrict__ kb,
                 const u16b* __restrict__ vT, const u16b* __restrict__ qps,
                 const u16b* __restrict__ kpst, u16b* __restrict__ ctx) {
  __shared__ __align__(16) u16b ptile[16 * 32];
  const int lane = threadIdx.x & 31;
  const int half = lane >> 4, nl = lane & 15;
  const int q0 = blockIdx.x * 16;
  const int h  = blockIdx.y;
  const int b  = blockIdx.z;

  const u16b* qbase = qb + ((long long)(b * S_ + q0)) * H_ + h * HD_;
  v16bf aq0 = frag_a(qbase,      H_, lane);   // dims 0..31
  v16bf aq1 = frag_a(qbase + 32, H_, lane);   // dims 32..63

  const u16b* kbz   = kb   + (long long)b * S_ * H_ + h * HD_;
  const u16b* vTz   = vT   + ((long long)(b * NH_ + h)) * HD_ * S_;
  const u16b* qpsz  = qps  + ((long long)(b * NH_ + h)) * S_ * WIN_;
  const u16b* kpstz = kpst + ((long long)(b * NH_ + h)) * S_ * WIN_;

  v8f O[4] = {};
  float mrun[8], lrun[8];
#pragma unroll
  for (int i = 0; i < 8; ++i) { mrun[i] = -1e30f; lrun[i] = 0.f; }

  for (int k0 = 0; k0 < q0 + 16; k0 += 32) {
    // scores for two 16-key subtiles
    v8f s[2] = {};
#pragma unroll
    for (int t = 0; t < 2; ++t) {
      const u16b* kt = kbz + (long long)(k0 + t * 16) * H_;
      v16bf b0 = frag_b(kt,      H_, lane);
      v16bf b1 = frag_b(kt + 32, H_, lane);
      s[t] = wmma_bf16(aq0, b0, s[t]);
      s[t] = wmma_bf16(aq1, b1, s[t]);
    }
    // issue V-tile loads now: in flight across softmax VALU + LDS round-trip
    v16bf vfr[4];
#pragma unroll
    for (int t2 = 0; t2 < 4; ++t2)
      vfr[t2] = frag_b(vTz + (long long)(t2 * 16) * S_ + k0, S_, lane);

    float p[2][8], corr[8];
#pragma unroll
    for (int i = 0; i < 8; ++i) {
      int m = q0 + i + 8 * half;
      float sv[2];
      float rmax = -1e30f;
#pragma unroll
      for (int t = 0; t < 2; ++t) {
        int n = k0 + t * 16 + nl;
        int rel = n - m + (WIN_ - 1);
        rel = rel < 0 ? 0 : (rel > WIN_ - 1 ? WIN_ - 1 : rel);   // branch-free, in-bounds
        float g = bf2f(qpsz[(long long)m * WIN_ + rel]) +
                  bf2f(kpstz[(long long)n * WIN_ + rel]);
        float val = (s[t][i] + g) * INV_SCALE;
        val = (n > m) ? -1e30f : val;                            // causal mask via select
        sv[t] = val;
        rmax = fmaxf(rmax, val);
      }
#pragma unroll
      for (int msk = 8; msk >= 1; msk >>= 1)
        rmax = fmaxf(rmax, __shfl_xor(rmax, msk, 32));
      float mnew = fmaxf(mrun[i], rmax);
      corr[i] = __expf(mrun[i] - mnew);
      mrun[i] = mnew;
      float rsum = 0.f;
#pragma unroll
      for (int t = 0; t < 2; ++t) { p[t][i] = __expf(sv[t] - mnew); rsum += p[t][i]; }
#pragma unroll
      for (int msk = 8; msk >= 1; msk >>= 1)
        rsum += __shfl_xor(rsum, msk, 32);
      lrun[i] = lrun[i] * corr[i] + rsum;
    }
#pragma unroll
    for (int t2 = 0; t2 < 4; ++t2)
#pragma unroll
      for (int i = 0; i < 8; ++i) O[t2][i] *= corr[i];

    // P (D layout) -> LDS bf16 -> A fragment (16 queries x 32 keys)
    __syncthreads();
#pragma unroll
    for (int i = 0; i < 8; ++i) {
      ptile[(i + 8 * half) * 32 + nl]      = f2bf(p[0][i]);
      ptile[(i + 8 * half) * 32 + 16 + nl] = f2bf(p[1][i]);
    }
    __syncthreads();
    v16bf pa = frag_a(ptile, 32, lane);
#pragma unroll
    for (int t2 = 0; t2 < 4; ++t2)
      O[t2] = wmma_bf16(pa, vfr[t2], O[t2]);
  }
  // epilogue: normalize, store bf16 context
  u16b* cp = ctx + ((long long)(b * S_ + q0)) * H_ + h * HD_;
#pragma unroll
  for (int i = 0; i < 8; ++i) {
    float invl = 1.f / lrun[i];
#pragma unroll
    for (int t2 = 0; t2 < 4; ++t2)
      cp[(long long)(i + 8 * half) * H_ + t2 * 16 + nl] = f2bf(O[t2][i] * invl);
  }
}

// ---------------- launch ----------------
extern "C" void kernel_launch(void* const* d_in, const int* in_sizes, int n_in,
                              void* d_out, int out_size, void* d_ws, size_t ws_size,
                              hipStream_t stream) {
  (void)in_sizes; (void)n_in; (void)out_size; (void)ws_size;
  const float* x     = (const float*)d_in[0];
  const float* Wq    = (const float*)d_in[1];
  const float* bq    = (const float*)d_in[2];
  const float* Wk    = (const float*)d_in[3];
  const float* bk    = (const float*)d_in[4];
  const float* Wv    = (const float*)d_in[5];
  const float* bv    = (const float*)d_in[6];
  const float* Wo    = (const float*)d_in[7];
  const float* bo    = (const float*)d_in[8];
  const float* tbl   = (const float*)d_in[9];
  const float* gamma = (const float*)d_in[10];
  const float* beta  = (const float*)d_in[11];
  const float* Wrk   = (const float*)d_in[12];
  const float* brk   = (const float*)d_in[13];
  const float* Wrq   = (const float*)d_in[14];
  const float* brq   = (const float*)d_in[15];
  float* out = (float*)d_out;

  char* ws = (char*)d_ws;
  size_t cur = 0;
  auto alloc = [&](size_t bytes) -> void* {
    void* p = ws + cur;
    cur = (cur + bytes + 255) & ~(size_t)255;
    return p;
  };
  const size_t NXE = (size_t)B_ * S_ * H_;      // 4M elems
  const size_t NWE = (size_t)H_ * H_;           // 1M elems
  u16b* xb    = (u16b*)alloc(NXE * 2);
  u16b* WqT   = (u16b*)alloc(NWE * 2);
  u16b* WkT   = (u16b*)alloc(NWE * 2);
  u16b* WvT   = (u16b*)alloc(NWE * 2);
  u16b* WoT   = (u16b*)alloc(NWE * 2);
  u16b* WrkT  = (u16b*)alloc(NWE * 2);
  u16b* WrqT  = (u16b*)alloc(NWE * 2);
  u16b* eb    = (u16b*)alloc((size_t)WIN_ * H_ * 2);
  u16b* qb    = (u16b*)alloc(NXE * 2);
  u16b* kb    = (u16b*)alloc(NXE * 2);
  u16b* vb    = (u16b*)alloc(NXE * 2);
  u16b* rkb   = (u16b*)alloc((size_t)WIN_ * H_ * 2);
  u16b* rqb   = (u16b*)alloc((size_t)WIN_ * H_ * 2);
  u16b* vTt   = (u16b*)alloc(NXE * 2);
  u16b* qps   = (u16b*)alloc((size_t)B_ * NH_ * S_ * WIN_ * 2);
  u16b* kpst  = (u16b*)alloc((size_t)B_ * NH_ * S_ * WIN_ * 2);
  u16b* ctxb  = (u16b*)alloc(NXE * 2);

  // 1) conversions (coalesced / tiled)
  f2bf_kernel<<<(int)(NXE / 4 / 256), 256, 0, stream>>>(x, xb, (int)(NXE / 4));
  dim3 gW(H_ / 32, H_ / 32, 1);
  wtrans_kernel<<<gW, 256, 0, stream>>>(Wq,  WqT);
  wtrans_kernel<<<gW, 256, 0, stream>>>(Wk,  WkT);
  wtrans_kernel<<<gW, 256, 0, stream>>>(Wv,  WvT);
  wtrans_kernel<<<gW, 256, 0, stream>>>(Wo,  WoT);
  wtrans_kernel<<<gW, 256, 0, stream>>>(Wrk, WrkT);
  wtrans_kernel<<<gW, 256, 0, stream>>>(Wrq, WrqT);
  lnorm_kernel<<<WIN_, 256, 0, stream>>>(tbl, gamma, beta, eb);

  // 2) projections: q/k/v = x@W + b   (M=4096, N=1024, K=1024)
  dim3 gP(B_ * S_ / 128, H_ / 64, 1);
  gemm_wmma<<<gP, 256, 0, stream>>>(xb, H_, 0, 0, WqT, H_, 0, 0, bq,
                                    qb, H_, 0, 0, 1, B_ * S_, H_, H_, 0);
  gemm_wmma<<<gP, 256, 0, stream>>>(xb, H_, 0, 0, WkT, H_, 0, 0, bk,
                                    kb, H_, 0, 0, 1, B_ * S_, H_, H_, 0);
  gemm_wmma<<<gP, 256, 0, stream>>>(xb, H_, 0, 0, WvT, H_, 0, 0, bv,
                                    vb, H_, 0, 0, 1, B_ * S_, H_, H_, 0);

  // 3) rel-pos projections: rk/rq = e@W + b  (M=512, N=1024, K=1024)
  dim3 gR(WIN_ / 128, H_ / 64, 1);
  gemm_wmma<<<gR, 256, 0, stream>>>(eb, H_, 0, 0, WrkT, H_, 0, 0, brk,
                                    rkb, H_, 0, 0, 1, WIN_, H_, H_, 0);
  gemm_wmma<<<gR, 256, 0, stream>>>(eb, H_, 0, 0, WrqT, H_, 0, 0, brq,
                                    rqb, H_, 0, 0, 1, WIN_, H_, H_, 0);

  // 4) v transpose to [b,h,d,S]
  vtrans_kernel<<<dim3(S_ / 32, HD_ / 32, B_ * NH_), 256, 0, stream>>>(vb, vTt);

  // 5) qps[b,h,q,w] = q . rk^T ; kpst[b,h,k,w] = k . rq^T   (batched per (b,h))
  dim3 gS(S_ / 128, WIN_ / 64, B_ * NH_);
  gemm_wmma<<<gS, 256, 0, stream>>>(qb, H_, (long long)S_ * H_, HD_,
                                    rkb, H_, 0, HD_, nullptr,
                                    qps, WIN_, (long long)NH_ * S_ * WIN_,
                                    (long long)S_ * WIN_, NH_, S_, WIN_, HD_, 0);
  gemm_wmma<<<gS, 256, 0, stream>>>(kb, H_, (long long)S_ * H_, HD_,
                                    rqb, H_, 0, HD_, nullptr,
                                    kpst, WIN_, (long long)NH_ * S_ * WIN_,
                                    (long long)S_ * WIN_, NH_, S_, WIN_, HD_, 0);

  // 6) attention
  attn_kernel<<<dim3(S_ / 16, NH_, B_), 32, 0, stream>>>(qb, kb, vTt, qps, kpst, ctxb);

  // 7) output projection: out = ctx@Wo + bo  (fp32 out)
  gemm_wmma<<<gP, 256, 0, stream>>>(ctxb, H_, 0, 0, WoT, H_, 0, 0, bo,
                                    out, H_, 0, 0, 1, B_ * S_, H_, H_, 1);
}
